// DSSM_71949292143295
// MI455X (gfx1250) — matrix operand order repaired and checked by reading, not compiled, not run on hardware
//
#include <hip/hip_runtime.h>
#include <hip/hip_bf16.h>
#include <math.h>

// ---------------------------------------------------------------------------
// DSSM for MI455X (gfx1250). All matmuls use v_wmma_f32_16x16x32_f16.
// Batched GEMMs stage the B-panel into LDS with global_load_async_to_lds_b128
// (double-buffered, ASYNCcnt-synchronized). Sequence tensors are [T, B, D]
// ("scan-major") so the GEMM A-matrix is plain row-major with row = t*B + b.
// ---------------------------------------------------------------------------

typedef _Float16 half_t;
typedef __attribute__((ext_vector_type(16))) _Float16 v16h;
typedef __attribute__((ext_vector_type(8)))  float    v8f;

#define B_   32
#define T_   512
#define DOBS 64
#define DLAT 128
#define DHID 512
#define BT_  (B_ * T_)   // 16384

// ---------------- device helpers ----------------

__device__ __forceinline__ float sigf(float x) { return 1.0f / (1.0f + __expf(-x)); }
__device__ __forceinline__ float softplusf(float x) {
  return (x > 20.0f) ? x : log1pf(__expf(x));
}

__device__ __forceinline__ v8f wmma32(v16h a, v16h b, v8f c) {
  // (neg_a, A, neg_b, B, c_mod, C, reuse_a, reuse_b)
  return __builtin_amdgcn_wmma_f32_16x16x32_f16(false, a, false, b, (short)0, c,
                                                false, false);
}

// A fragment: 16x32 (MxK) tile from row-major A (caller passes A + r0*lda + k0).
// Layout: lanes 0-15 -> M=lane, K = 0..15; lanes 16-31 -> K = 16..31.
__device__ __forceinline__ v16h load_a_frag(const half_t* __restrict__ A, int lda,
                                            int lane) {
  int m  = lane & 15;
  int kb = (lane & 16) ? 16 : 0;
  const half_t* p = A + (size_t)m * lda + kb;
  v16h a;
#pragma unroll
  for (int e = 0; e < 8; ++e) a[e] = p[e];
#pragma unroll
  for (int e = 0; e < 8; ++e) a[8 + e] = p[8 + e];
  return a;
}

// B fragment: logical B[k][n] = W[n0+n][k0+k]  (W row-major [N x K]).
// Caller passes W + n0*ldw + k0. Lanes 0-15: K=0..15, lanes 16-31: K=16..31.
__device__ __forceinline__ v16h load_b_frag(const half_t* __restrict__ W, int ldw,
                                            int lane) {
  int n  = lane & 15;
  int kb = (lane & 16) ? 16 : 0;
  const half_t* q = W + (size_t)n * ldw + kb;
  v16h b;
#pragma unroll
  for (int e = 0; e < 16; ++e) b[e] = q[e];
  return b;
}

// B fragment from an LDS panel laid out [n][k] with ldk = 32 halves.
__device__ __forceinline__ v16h load_b_frag_lds(const half_t* p, int lane) {
  int n  = lane & 15;
  int kb = (lane & 16) ? 16 : 0;
  const half_t* q = p + n * 32 + kb;
  v16h b;
#pragma unroll
  for (int e = 0; e < 16; ++e) b[e] = q[e];
  return b;
}

// ---------------- utility kernels ----------------

__global__ void k_cvt_f16(const float* __restrict__ in, half_t* __restrict__ out,
                          int n) {
  int i = blockIdx.x * blockDim.x + threadIdx.x;
  if (i < n) out[i] = (half_t)in[i];
}

// x [B,T,64] f32 -> xT16 [T,B,64] f16
__global__ void k_tx_cvt(const float* __restrict__ x, half_t* __restrict__ out) {
  int i = blockIdx.x * blockDim.x + threadIdx.x;
  if (i >= B_ * T_ * DOBS) return;
  int d = i % DOBS;
  int t = (i / DOBS) % T_;
  int b = i / (DOBS * T_);
  out[((size_t)t * B_ + b) * DOBS + d] = (half_t)x[i];
}

__global__ void k_zero_f32(float* __restrict__ p, int n) {
  int i = blockIdx.x * blockDim.x + threadIdx.x;
  if (i < n) p[i] = 0.0f;
}

__global__ void k_zero_f16(half_t* __restrict__ p, int n) {
  int i = blockIdx.x * blockDim.x + threadIdx.x;
  if (i < n) p[i] = (half_t)0.0f;
}

// ---------------- WMMA GEMM:  C[M,N] = A[M,K] * W[N,K]^T + b0 + b1 ----------
// grid = (N/64, M/64), block = 128 (4 waves). Wave w computes a 16x64 strip
// (rows r0 = by*64 + w*16, cols n0..n0+63) with 4 accumulators reusing one
// A-fragment per K-step. The 64x32 B-panel is staged into LDS with async
// loads (global_load_async_to_lds_b128), double-buffered on ASYNCcnt.
__global__ void k_gemm(const half_t* __restrict__ A, int lda,
                       const half_t* __restrict__ W, int ldw,
                       const float* __restrict__ b0, const float* __restrict__ b1,
                       float* __restrict__ C, int ldc, int K) {
  __shared__ half_t bs[2][64 * 32];  // [buf][n * 32 + k], 4 KB per buffer
  int lane = threadIdx.x & 31;
  int wave = threadIdx.x >> 5;
  int n0 = blockIdx.x * 64;
  int r0 = blockIdx.y * 64 + wave * 16;

  // async-stage 64 rows x 32 halves of W (rows n0..n0+63, cols k0..k0+31):
  // 256 x 16B units, 2 per thread.
  auto stage = [&](int buf, int k0) {
    int u0 = threadIdx.x * 2;
#pragma unroll
    for (int u = u0; u < u0 + 2; ++u) {
      int row = u >> 2;            // 4 units (64B) per row
      int ko  = (u & 3) * 8;       // 8 halves per unit
      const half_t* g = W + (size_t)(n0 + row) * ldw + k0 + ko;
      unsigned int lo = (unsigned int)(size_t)&bs[buf][row * 32 + ko];
      asm volatile("global_load_async_to_lds_b128 %0, %1, off"
                   :: "v"(lo), "v"(g) : "memory");
    }
  };

  v8f acc[4] = {{}, {}, {}, {}};
  stage(0, 0);
  asm volatile("s_wait_asynccnt 0x0" ::: "memory");
  __syncthreads();

  int nk = K / 32;
  for (int i = 0; i < nk; ++i) {
    int buf = i & 1;
    if (i + 1 < nk) stage(buf ^ 1, (i + 1) * 32);
    __builtin_prefetch(A + (size_t)r0 * lda + (i + 2) * 32, 0, 1);
    v16h af = load_a_frag(A + (size_t)r0 * lda + i * 32, lda, lane);
#pragma unroll
    for (int ct = 0; ct < 4; ++ct) {
      v16h bf = load_b_frag_lds(&bs[buf][ct * 16 * 32], lane);
      acc[ct] = wmma32(af, bf, acc[ct]);
    }
    asm volatile("s_wait_asynccnt 0x0" ::: "memory");
    __syncthreads();
  }

  int nb    = n0 + (lane & 15);
  int mbase = r0 + ((lane & 16) ? 8 : 0);
#pragma unroll
  for (int ct = 0; ct < 4; ++ct) {
    int n = nb + ct * 16;
    float bias = (b0 ? b0[n] : 0.0f) + (b1 ? b1[n] : 0.0f);
#pragma unroll
    for (int r = 0; r < 8; ++r)
      C[(size_t)(mbase + r) * ldc + n] = acc[ct][r] + bias;
  }
}

// ---------------- fused LSTM step ----------------
// grid = 32 (H/16), block = 256 (8 waves). Wave w: rt=w&1 row tile, gate=w>>1.
// g = xg_t + h_prev @ Whh^T; then i/f/g/o pointwise + c,h update.
__global__ void k_lstm_step(const float* __restrict__ xg_t,    // [32 x 2048]
                            const half_t* __restrict__ hprev,  // [32 x 512] f16
                            const half_t* __restrict__ Whh,    // [2048 x 512] f16
                            float* __restrict__ c,             // [32 x 512] in/out
                            half_t* __restrict__ hout) {       // [32 x 512] f16
  __shared__ float g[4][B_][16];
  int lane = threadIdx.x & 31;
  int wave = threadIdx.x >> 5;
  int rt = wave & 1, gate = wave >> 1;
  int col0 = gate * DHID + blockIdx.x * 16;
  int r0 = rt * 16;
  int n = lane & 15;
  int moff = (lane & 16) ? 8 : 0;
  v8f acc;
#pragma unroll
  for (int r = 0; r < 8; ++r)
    acc[r] = xg_t[(size_t)(r0 + moff + r) * (4 * DHID) + col0 + n];
  for (int k0 = 0; k0 < DHID; k0 += 32) {
    v16h af = load_a_frag(hprev + (size_t)r0 * DHID + k0, DHID, lane);
    v16h bf = load_b_frag(Whh + (size_t)col0 * DHID + k0, DHID, lane);
    acc = wmma32(af, bf, acc);
  }
#pragma unroll
  for (int r = 0; r < 8; ++r) g[gate][r0 + moff + r][n] = acc[r];
  __syncthreads();
  int j0 = blockIdx.x * 16;
  for (int i = threadIdx.x; i < B_ * 16; i += 256) {
    int b = i >> 4, jj = i & 15, j = j0 + jj;
    float gi = g[0][b][jj], gf = g[1][b][jj], gg = g[2][b][jj], go = g[3][b][jj];
    float cp = c[b * DHID + j];
    float cn = sigf(gf) * cp + sigf(gi) * tanhf(gg);
    float h  = sigf(go) * tanhf(cn);
    c[b * DHID + j]    = cn;
    hout[b * DHID + j] = (half_t)h;
  }
}

// ---------------- fused GRU prior step ----------------
// grid = 8 (L/16), block = 192 (6 waves). hg = h@Whh^T + bhh; r,u,n gates.
__global__ void k_gru_step(const float* __restrict__ xg_t,     // [32 x 384]
                           const half_t* __restrict__ hprev16, // [32 x 128] f16
                           const float* __restrict__ hprev32,  // [32 x 128] f32
                           const half_t* __restrict__ Whh,     // [384 x 128] f16
                           const float* __restrict__ bhh,      // [384]
                           float* __restrict__ hout32,         // [32 x 128]
                           half_t* __restrict__ hout16) {      // [32 x 128]
  __shared__ float g[3][B_][16];
  int lane = threadIdx.x & 31;
  int wave = threadIdx.x >> 5;
  int rt = wave & 1, gate = wave >> 1;
  int col0 = gate * DLAT + blockIdx.x * 16;
  int r0 = rt * 16;
  int n = lane & 15;
  int moff = (lane & 16) ? 8 : 0;
  float bv = bhh[col0 + n];
  v8f acc;
#pragma unroll
  for (int r = 0; r < 8; ++r) acc[r] = bv;
  for (int k0 = 0; k0 < DLAT; k0 += 32) {
    v16h af = load_a_frag(hprev16 + (size_t)r0 * DLAT + k0, DLAT, lane);
    v16h bf = load_b_frag(Whh + (size_t)col0 * DLAT + k0, DLAT, lane);
    acc = wmma32(af, bf, acc);
  }
#pragma unroll
  for (int r = 0; r < 8; ++r) g[gate][r0 + moff + r][n] = acc[r];
  __syncthreads();
  int j0 = blockIdx.x * 16;
  for (int i = threadIdx.x; i < B_ * 16; i += 192) {
    int b = i >> 4, jj = i & 15, j = j0 + jj;
    float xr = xg_t[b * (3 * DLAT) + j];
    float xu = xg_t[b * (3 * DLAT) + DLAT + j];
    float xn = xg_t[b * (3 * DLAT) + 2 * DLAT + j];
    float hr = g[0][b][jj], hu = g[1][b][jj], hn = g[2][b][jj];
    float hp = hprev32[b * DLAT + j];
    float r = sigf(xr + hr);
    float u = sigf(xu + hu);
    float nn = tanhf(xn + r * hn);
    float h = (1.0f - u) * nn + u * hp;
    hout32[b * DLAT + j] = h;
    hout16[b * DLAT + j] = (half_t)h;
  }
}

// ---------------- z = mu + exp(0.5 lv) * eps ; write z to d_out[1:] --------
__global__ void k_zsample(const float* __restrict__ mu, const float* __restrict__ lv,
                          const float* __restrict__ eps, float* __restrict__ out,
                          half_t* __restrict__ z16) {
  int i = blockIdx.x * blockDim.x + threadIdx.x;
  if (i >= BT_ * DLAT) return;
  int l = i & (DLAT - 1);
  int row = i >> 7;            // t*B + b
  int b = row & (B_ - 1);
  int t = row >> 5;
  float z = mu[i] + __expf(0.5f * lv[i]) * eps[((size_t)b * T_ + t) * DLAT + l];
  out[1 + ((size_t)b * T_ + t) * DLAT + l] = z;  // [B,T,L] output order
  z16[i] = (half_t)z;
}

// ---------------- silu in-place + f16 copy ----------------
__global__ void k_silu(float* __restrict__ io, half_t* __restrict__ o16, int n) {
  int i = blockIdx.x * blockDim.x + threadIdx.x;
  if (i >= n) return;
  float v = io[i];
  float s = v * sigf(v);
  io[i] = s;
  o16[i] = (half_t)s;
}

// ---------------- LayerNorm(row) of (a+b), rows of 512, -> f16 -------------
__global__ void k_ln(const float* __restrict__ a, const float* __restrict__ b,
                     const float* __restrict__ gam, const float* __restrict__ bet,
                     half_t* __restrict__ out) {
  __shared__ float red[256];
  int row = blockIdx.x, tid = threadIdx.x;
  const float* pa = a + (size_t)row * DHID;
  const float* pb = b + (size_t)row * DHID;
  float v0 = pa[tid] + pb[tid];
  float v1 = pa[tid + 256] + pb[tid + 256];
  red[tid] = v0 + v1;
  __syncthreads();
  for (int s = 128; s; s >>= 1) { if (tid < s) red[tid] += red[tid + s]; __syncthreads(); }
  float mean = red[0] * (1.0f / DHID);
  __syncthreads();
  float d0 = v0 - mean, d1 = v1 - mean;
  red[tid] = d0 * d0 + d1 * d1;
  __syncthreads();
  for (int s = 128; s; s >>= 1) { if (tid < s) red[tid] += red[tid + s]; __syncthreads(); }
  float rstd = rsqrtf(red[0] * (1.0f / DHID) + 1e-5f);
  half_t* po = out + (size_t)row * DHID;
  po[tid]       = (half_t)(d0 * rstd * gam[tid] + bet[tid]);
  po[tid + 256] = (half_t)(d1 * rstd * gam[tid + 256] + bet[tid + 256]);
}

// ---------------- Student-t recon loss, sum/B into out[0] ------------------
__global__ void k_recon(const float* __restrict__ dp, const float* __restrict__ x,
                        float* __restrict__ out) {
  int i = blockIdx.x * blockDim.x + threadIdx.x;
  float val = 0.0f;
  if (i < BT_ * DOBS) {
    int d = i & (DOBS - 1);
    int row = i >> 6;                 // t*B + b
    int b = row & (B_ - 1);
    int t = row >> 5;
    float xv = x[((size_t)b * T_ + t) * DOBS + d];
    float mu = dp[(size_t)row * (3 * DOBS) + d];
    float sc = softplusf(dp[(size_t)row * (3 * DOBS) + DOBS + d]) + 1e-4f;
    float df = softplusf(dp[(size_t)row * (3 * DOBS) + 2 * DOBS + d]) + 2.0f;
    float tt = (xv - mu) / sc;
    float lp = lgammaf(0.5f * (df + 1.0f)) - lgammaf(0.5f * df)
             - 0.5f * logf(df * 3.14159265358979323846f) - logf(sc)
             - 0.5f * (df + 1.0f) * log1pf(tt * tt / df);
    val = -lp * (1.0f / B_);
  }
  __shared__ float red[256];
  red[threadIdx.x] = val;
  __syncthreads();
  for (int s = 128; s; s >>= 1) { if (threadIdx.x < s) red[threadIdx.x] += red[threadIdx.x + s]; __syncthreads(); }
  if (threadIdx.x == 0) atomicAdd(out, red[0]);
}

// ---------------- KL( N(mu,std) || N(zprior,1) ), sum/B into out[0] --------
__global__ void k_kl(const float* __restrict__ mu, const float* __restrict__ lv,
                     const float* __restrict__ zp, float* __restrict__ out) {
  int i = blockIdx.x * blockDim.x + threadIdx.x;
  float val = 0.0f;
  if (i < BT_ * DLAT) {
    float l = lv[i];
    float s2 = __expf(l);
    float dm = mu[i] - zp[i];
    val = (-0.5f * l + 0.5f * (s2 + dm * dm) - 0.5f) * (1.0f / B_);
  }
  __shared__ float red[256];
  red[threadIdx.x] = val;
  __syncthreads();
  for (int s = 128; s; s >>= 1) { if (threadIdx.x < s) red[threadIdx.x] += red[threadIdx.x + s]; __syncthreads(); }
  if (threadIdx.x == 0) atomicAdd(out, red[0]);
}

// ---------------------------------------------------------------------------

extern "C" void kernel_launch(void* const* d_in, const int* in_sizes, int n_in,
                              void* d_out, int out_size, void* d_ws, size_t ws_size,
                              hipStream_t stream) {
  (void)in_sizes; (void)n_in; (void)out_size; (void)ws_size;

  const float* x   = (const float*)d_in[0];
  const float* eps = (const float*)d_in[1];
  const float* Wih[3] = {(const float*)d_in[2], (const float*)d_in[6], (const float*)d_in[10]};
  const float* Whh[3] = {(const float*)d_in[3], (const float*)d_in[7], (const float*)d_in[11]};
  const float* bih[3] = {(const float*)d_in[4], (const float*)d_in[8], (const float*)d_in[12]};
  const float* bhh[3] = {(const float*)d_in[5], (const float*)d_in[9], (const float*)d_in[13]};
  const float* mu_W = (const float*)d_in[14]; const float* mu_b = (const float*)d_in[15];
  const float* lv_W = (const float*)d_in[16]; const float* lv_b = (const float*)d_in[17];
  const float* gWih = (const float*)d_in[18]; const float* gWhh = (const float*)d_in[19];
  const float* gbih = (const float*)d_in[20]; const float* gbhh = (const float*)d_in[21];
  const float* dW1 = (const float*)d_in[22]; const float* db1 = (const float*)d_in[23];
  const float* rW1 = (const float*)d_in[24]; const float* rb1 = (const float*)d_in[25];
  const float* rW2 = (const float*)d_in[26]; const float* rb2 = (const float*)d_in[27];
  const float* lng = (const float*)d_in[28]; const float* lnb = (const float*)d_in[29];
  const float* dW2 = (const float*)d_in[30]; const float* db2 = (const float*)d_in[31];
  float* out = (float*)d_out;

  // ---- workspace carve-up ----
  char* base = (char*)d_ws;
  size_t off = 0;
  auto alloc = [&](size_t bytes) -> char* {
    char* r = base + off;
    off += (bytes + 255) & ~(size_t)255;
    return r;
  };
  half_t* xT16  = (half_t*)alloc((size_t)BT_ * DOBS * 2);
  half_t* hseqA = (half_t*)alloc((size_t)BT_ * DHID * 2);
  half_t* hseqB = (half_t*)alloc((size_t)BT_ * DHID * 2);
  half_t* wIh16[3], *wHh16[3];
  for (int l = 0; l < 3; ++l) {
    int din = l ? DHID : DOBS;
    wIh16[l] = (half_t*)alloc((size_t)4 * DHID * din * 2);
    wHh16[l] = (half_t*)alloc((size_t)4 * DHID * DHID * 2);
  }
  half_t* wMu16 = (half_t*)alloc((size_t)DLAT * DHID * 2);
  half_t* wLv16 = (half_t*)alloc((size_t)DLAT * DHID * 2);
  half_t* wGih16 = (half_t*)alloc((size_t)3 * DLAT * DLAT * 2);
  half_t* wGhh16 = (half_t*)alloc((size_t)3 * DLAT * DLAT * 2);
  half_t* wD116 = (half_t*)alloc((size_t)DHID * DLAT * 2);
  half_t* wR116 = (half_t*)alloc((size_t)DHID * DHID * 2);
  half_t* wR216 = (half_t*)alloc((size_t)DHID * DHID * 2);
  half_t* wD216 = (half_t*)alloc((size_t)3 * DOBS * DHID * 2);
  float* xg32   = (float*)alloc((size_t)BT_ * 4 * DHID * 4);   // 128 MiB, reused
  float* cbuf   = (float*)alloc((size_t)B_ * DHID * 4);
  half_t* zero16 = (half_t*)alloc((size_t)B_ * DHID * 2);
  float* zero32  = (float*)alloc((size_t)B_ * DHID * 4);
  float* mu32   = (float*)alloc((size_t)BT_ * DLAT * 4);
  float* lv32   = (float*)alloc((size_t)BT_ * DLAT * 4);
  half_t* z16   = (half_t*)alloc((size_t)BT_ * DLAT * 2);
  float* zprior = (float*)alloc((size_t)BT_ * DLAT * 4);
  half_t* gh16  = (half_t*)alloc((size_t)BT_ * DLAT * 2);
  half_t* hd16  = (half_t*)alloc((size_t)BT_ * DHID * 2);
  half_t* r116  = (half_t*)alloc((size_t)BT_ * DHID * 2);
  half_t* hd216 = (half_t*)alloc((size_t)BT_ * DHID * 2);
  float* dp32   = (float*)alloc((size_t)BT_ * 3 * DOBS * 4);
  // regions overlaid inside xg32 (free after LSTM phase)
  float* gruxg = xg32;                                               // BT*384
  float* hd32  = (float*)((char*)xg32 + ((size_t)32 << 20));         // BT*512
  float* r132  = (float*)((char*)xg32 + ((size_t)64 << 20));         // BT*512
  float* res32 = (float*)((char*)xg32 + ((size_t)96 << 20));         // BT*512

  auto cvt = [&](const float* s, half_t* d, int n) {
    k_cvt_f16<<<(n + 255) / 256, 256, 0, stream>>>(s, d, n);
  };
  auto gemm = [&](const half_t* A, int lda, const half_t* W, int ldw,
                  const float* b0, const float* b1, float* C, int M, int N, int K) {
    dim3 g(N / 64, M / 64);
    k_gemm<<<g, 128, 0, stream>>>(A, lda, W, ldw, b0, b1, C, N, K);
  };

  // ---- init ----
  k_zero_f32<<<1, 32, 0, stream>>>(out, 1);
  k_zero_f16<<<(B_ * DHID + 255) / 256, 256, 0, stream>>>(zero16, B_ * DHID);
  k_zero_f32<<<(B_ * DHID + 255) / 256, 256, 0, stream>>>(zero32, B_ * DHID);

  // ---- weight conversion (f32 -> f16) ----
  for (int l = 0; l < 3; ++l) {
    int din = l ? DHID : DOBS;
    cvt(Wih[l], wIh16[l], 4 * DHID * din);
    cvt(Whh[l], wHh16[l], 4 * DHID * DHID);
  }
  cvt(mu_W, wMu16, DLAT * DHID);
  cvt(lv_W, wLv16, DLAT * DHID);
  cvt(gWih, wGih16, 3 * DLAT * DLAT);
  cvt(gWhh, wGhh16, 3 * DLAT * DLAT);
  cvt(dW1, wD116, DHID * DLAT);
  cvt(rW1, wR116, DHID * DHID);
  cvt(rW2, wR216, DHID * DHID);
  cvt(dW2, wD216, 3 * DOBS * DHID);
  k_tx_cvt<<<(BT_ * DOBS + 255) / 256, 256, 0, stream>>>(x, xT16);

  // ---- 3-layer LSTM encoder ----
  half_t* hin = nullptr;
  half_t* hcur = hseqA;
  for (int l = 0; l < 3; ++l) {
    const half_t* A = (l == 0) ? xT16 : hin;
    int K = (l == 0) ? DOBS : DHID;
    gemm(A, K, wIh16[l], K, bih[l], bhh[l], xg32, BT_, 4 * DHID, K);
    k_zero_f32<<<(B_ * DHID + 255) / 256, 256, 0, stream>>>(cbuf, B_ * DHID);
    for (int t = 0; t < T_; ++t) {
      const half_t* hp = t ? hcur + (size_t)(t - 1) * B_ * DHID : zero16;
      k_lstm_step<<<DHID / 16, 256, 0, stream>>>(
          xg32 + (size_t)t * B_ * 4 * DHID, hp, wHh16[l], cbuf,
          hcur + (size_t)t * B_ * DHID);
    }
    hin = hcur;
    hcur = (hcur == hseqA) ? hseqB : hseqA;
  }

  // ---- mu / logvar heads, reparameterized z (output #2) ----
  gemm(hin, DHID, wMu16, DHID, mu_b, nullptr, mu32, BT_, DLAT, DHID);
  gemm(hin, DHID, wLv16, DHID, lv_b, nullptr, lv32, BT_, DLAT, DHID);
  k_zsample<<<(BT_ * DLAT + 255) / 256, 256, 0, stream>>>(mu32, lv32, eps, out, z16);

  // ---- GRU prior over z ----
  gemm(z16, DLAT, wGih16, DLAT, gbih, nullptr, gruxg, BT_, 3 * DLAT, DLAT);
  for (int t = 0; t < T_; ++t) {
    const half_t* hp16 = t ? gh16 + (size_t)(t - 1) * B_ * DLAT : zero16;
    const float* hp32 = t ? zprior + (size_t)(t - 1) * B_ * DLAT : zero32;
    k_gru_step<<<DLAT / 16, 192, 0, stream>>>(
        gruxg + (size_t)t * B_ * 3 * DLAT, hp16, hp32, wGhh16, gbhh,
        zprior + (size_t)t * B_ * DLAT, gh16 + (size_t)t * B_ * DLAT);
  }

  // ---- decoder: Linear -> SiLU -> ResBlock -> LN -> Linear ----
  gemm(z16, DLAT, wD116, DLAT, db1, nullptr, hd32, BT_, DHID, DLAT);
  k_silu<<<(BT_ * DHID + 255) / 256, 256, 0, stream>>>(hd32, hd16, BT_ * DHID);
  gemm(hd16, DHID, wR116, DHID, rb1, nullptr, r132, BT_, DHID, DHID);
  k_silu<<<(BT_ * DHID + 255) / 256, 256, 0, stream>>>(r132, r116, BT_ * DHID);
  gemm(r116, DHID, wR216, DHID, rb2, nullptr, res32, BT_, DHID, DHID);
  k_ln<<<BT_, 256, 0, stream>>>(hd32, res32, lng, lnb, hd216);
  gemm(hd216, DHID, wD216, DHID, db2, nullptr, dp32, BT_, 3 * DOBS, DHID);

  // ---- losses (output #1) ----
  k_recon<<<(BT_ * DOBS + 255) / 256, 256, 0, stream>>>(dp32, x, out);
  k_kl<<<(BT_ * DLAT + 255) / 256, 256, 0, stream>>>(mu32, lv32, zprior, out);
}
